// CoarseBlockAttention_44375602102971
// MI455X (gfx1250) — matrix operand order repaired and checked by reading, not compiled, not run on hardware
//
#include <hip/hip_runtime.h>
#include <hip/hip_bf16.h>
#include <math.h>

// ---------------------------------------------------------------------------
// CoarseBlockAttention, MI455X (gfx1250, wave32, WMMA).
// Memory-bound problem: 67MB read of x + 67MB write of out dominate
// (~5.8us at 23.3 TB/s).  All GEMMs use v_wmma_f32_16x16x32_f16.
// Strength reduction: Vsum = (sum_16 x) @ Wv^T + 16*bv  (16x fewer V FLOPs).
// ---------------------------------------------------------------------------

typedef __attribute__((ext_vector_type(16))) _Float16 v16h;
typedef __attribute__((ext_vector_type(8)))  _Float16 v8h;
typedef __attribute__((ext_vector_type(8)))  float    v8f;

// A-matrix 16x32 f16 fragment: lane<16 -> K=0..15 contiguous,
// lane>=16 -> K=16..31.  p = rowBase + k0 + (lane>>4)*16.
__device__ __forceinline__ v16h load_a_frag(const _Float16* p) {
    v8h lo = *(const v8h*)(p);
    v8h hi = *(const v8h*)(p + 8);
    return __builtin_shufflevector(lo, hi, 0,1,2,3,4,5,6,7,8,9,10,11,12,13,14,15);
}

// B-matrix 32x16 f16 fragment: lane<16 -> (N=lane, K=0..15 contiguous),
// lane>=16 -> K=16..31.  Since B[k][n] = Wrow[n][k], row-major weight rows
// load directly.  p = rowBase + k0 + (lane>>4)*16.
__device__ __forceinline__ v16h load_b_frag(const _Float16* p) {
    v8h lo = *(const v8h*)(p);
    v8h hi = *(const v8h*)(p + 8);
    return __builtin_shufflevector(lo, hi, 0,1,2,3,4,5,6,7,8,9,10,11,12,13,14,15);
}

// ---------------------------------------------------------------------------
// Kernel 0: f32 -> f16 conversion (weights).
__global__ void cvt_f16_kernel(const float* __restrict__ in,
                               _Float16* __restrict__ out, int n) {
    int i = blockIdx.x * blockDim.x + threadIdx.x;
    if (i < n) out[i] = (_Float16)in[i];
}

// ---------------------------------------------------------------------------
// Kernel 1: one 64x64 plane (b,c) per block.  Produces, in f16:
//   Xavg[(b*256+n)*512 + c] = mean of 4x4 spatial block n   (for Q,K)
//   Xsum[(b*256+n)*512 + c] = sum of 16 consecutive flat px (for Vsum)
__global__ void reduce_kernel(const float* __restrict__ x,
                              _Float16* __restrict__ xavg,
                              _Float16* __restrict__ xsum) {
    __shared__ float4 lds4[1024];                 // 16 KB plane
    float* lds = (float*)lds4;
    int plane = blockIdx.x;                       // b*512 + c
    int b = plane >> 9, c = plane & 511;
    const float4* p4 = (const float4*)(x + (size_t)plane * 4096);
    int t = threadIdx.x;                          // 0..255
#pragma unroll
    for (int i = 0; i < 4; ++i) lds4[t + 256 * i] = p4[t + 256 * i];
    __syncthreads();

    // 4x4 spatial block mean (block n = t: bh = t>>4, bw = t&15)
    int bh = t >> 4, bw = t & 15;
    float s = 0.f;
#pragma unroll
    for (int r = 0; r < 4; ++r)
#pragma unroll
        for (int cc = 0; cc < 4; ++cc)
            s += lds[(bh * 4 + r) * 64 + bw * 4 + cc];
    float avg = s * (1.f / 16.f);

    // flat 16-pixel sum (group n = t covers flat pixels [t*16, t*16+16))
    float fs = 0.f;
#pragma unroll
    for (int i = 0; i < 16; ++i) fs += lds[t * 16 + i];

    size_t o = ((size_t)(b * 256 + t)) * 512 + c;
    xavg[o] = (_Float16)avg;
    xsum[o] = (_Float16)fs;
}

// ---------------------------------------------------------------------------
// Kernel 2: out[m][n] = sum_k A[m][k]*W[n][k] + bias[n]*biasScale
// A: 2048x512 f16 row-major. W: 512x512 f16 row-major. One wave per 16x16 tile.
// transposeOut=0: out f16 row-major ld=512 (Q, K)
// transposeOut=1: out f16 as [b][n][m%256] ld=256 (VsumT, per-batch C x nb)
__global__ void qkv_gemm_kernel(const _Float16* __restrict__ A,
                                const _Float16* __restrict__ W,
                                const float* __restrict__ bias, float biasScale,
                                _Float16* __restrict__ out, int transposeOut) {
    int mt = blockIdx.x, nt = blockIdx.y;
    int lane = threadIdx.x & 31;
    int l16 = lane & 15, half = lane >> 4;
    const _Float16* Ap = A + (size_t)(mt * 16 + l16) * 512 + half * 16;
    const _Float16* Wp = W + (size_t)(nt * 16 + l16) * 512 + half * 16;
    v8f acc = {};
#pragma unroll 4
    for (int k0 = 0; k0 < 512; k0 += 32) {
        v16h a  = load_a_frag(Ap + k0);
        v16h bf = load_b_frag(Wp + k0);
        acc = __builtin_amdgcn_wmma_f32_16x16x32_f16(false, a, false, bf,
                                                     (short)0, acc, false, false);
    }
    int n = nt * 16 + l16;
    float bb = bias[n] * biasScale;
    if (!transposeOut) {
#pragma unroll
        for (int r = 0; r < 8; ++r)
            out[(size_t)(mt * 16 + r + 8 * half) * 512 + n] = (_Float16)(acc[r] + bb);
    } else {
        int b  = mt >> 4;                 // 16 m-tiles per batch
        int ml = (mt * 16) & 255;
#pragma unroll
        for (int r = 0; r < 8; ++r)
            out[(size_t)b * 512 * 256 + (size_t)n * 256 + (ml + r + 8 * half)] =
                (_Float16)(acc[r] + bb);
    }
}

// ---------------------------------------------------------------------------
// Kernel 3: scores[b][m][n] = (Q_b[m]·K_b[n]) / sqrt(512), f32.
__global__ void scores_kernel(const _Float16* __restrict__ Q,
                              const _Float16* __restrict__ K,
                              float* __restrict__ scores) {
    int mt = blockIdx.x, nt = blockIdx.y, b = blockIdx.z;
    int lane = threadIdx.x & 31;
    int l16 = lane & 15, half = lane >> 4;
    const _Float16* Qp = Q + (size_t)b * 256 * 512 + (size_t)(mt * 16 + l16) * 512 + half * 16;
    const _Float16* Kp = K + (size_t)b * 256 * 512 + (size_t)(nt * 16 + l16) * 512 + half * 16;
    v8f acc = {};
#pragma unroll 4
    for (int k0 = 0; k0 < 512; k0 += 32) {
        v16h a  = load_a_frag(Qp + k0);
        v16h bf = load_b_frag(Kp + k0);
        acc = __builtin_amdgcn_wmma_f32_16x16x32_f16(false, a, false, bf,
                                                     (short)0, acc, false, false);
    }
    const float scale = 0.044194173824159216f;    // 1/sqrt(512)
    float* outp = scores + (size_t)b * 65536 + (size_t)nt * 16 + l16;
#pragma unroll
    for (int r = 0; r < 8; ++r)
        outp[(size_t)(mt * 16 + r + 8 * half) * 256] = acc[r] * scale;
}

// ---------------------------------------------------------------------------
// Kernel 4: row softmax over 256, one wave per row, output f16.
__global__ void softmax_kernel(const float* __restrict__ scores,
                               _Float16* __restrict__ asoft) {
    int row  = blockIdx.x * (blockDim.x >> 5) + (threadIdx.x >> 5);  // 0..2047
    int lane = threadIdx.x & 31;
    const float* p = scores + (size_t)row * 256;
    float v[8], m = -INFINITY;
#pragma unroll
    for (int i = 0; i < 8; ++i) { v[i] = p[lane + 32 * i]; m = fmaxf(m, v[i]); }
#pragma unroll
    for (int off = 16; off >= 1; off >>= 1) m = fmaxf(m, __shfl_xor(m, off, 32));
    float s = 0.f;
#pragma unroll
    for (int i = 0; i < 8; ++i) { v[i] = __expf(v[i] - m); s += v[i]; }
#pragma unroll
    for (int off = 16; off >= 1; off >>= 1) s += __shfl_xor(s, off, 32);
    float inv = 1.f / s;
    _Float16* o = asoft + (size_t)row * 256;
#pragma unroll
    for (int i = 0; i < 8; ++i) o[lane + 32 * i] = (_Float16)(v[i] * inv);
}

// ---------------------------------------------------------------------------
// Kernel 5: out_small[b][m][c] = sum_n A_b[m][n] * VsumT_b[c][n], f32.
__global__ void av_kernel(const _Float16* __restrict__ Asoft,
                          const _Float16* __restrict__ VsumT,
                          float* __restrict__ outSmall) {
    int mt = blockIdx.x, ct = blockIdx.y, b = blockIdx.z;
    int lane = threadIdx.x & 31;
    int l16 = lane & 15, half = lane >> 4;
    const _Float16* Ap = Asoft + (size_t)b * 65536  + (size_t)(mt * 16 + l16) * 256 + half * 16;
    const _Float16* Bp = VsumT + (size_t)b * 131072 + (size_t)(ct * 16 + l16) * 256 + half * 16;
    v8f acc = {};
#pragma unroll
    for (int k0 = 0; k0 < 256; k0 += 32) {
        v16h a  = load_a_frag(Ap + k0);
        v16h bf = load_b_frag(Bp + k0);
        acc = __builtin_amdgcn_wmma_f32_16x16x32_f16(false, a, false, bf,
                                                     (short)0, acc, false, false);
    }
    float* outp = outSmall + (size_t)b * 131072 + (size_t)ct * 16 + l16;
#pragma unroll
    for (int r = 0; r < 8; ++r)
        outp[(size_t)(mt * 16 + r + 8 * half) * 512] = acc[r];
}

// ---------------------------------------------------------------------------
// Kernel 6: broadcast expand to (B,C,H,W): out[idx] = out_small[b][h*4+w/16][c]
__global__ void expand_kernel(const float* __restrict__ outSmall,
                              float* __restrict__ out) {
    int idx = blockIdx.x * 256 + threadIdx.x;     // b*2^21 + c*2^12 + h*2^6 + w
    int w = idx & 63, h = (idx >> 6) & 63;
    int c = (idx >> 12) & 511, b = idx >> 21;
    int n = h * 4 + (w >> 4);
    out[idx] = outSmall[((size_t)b * 256 + n) * 512 + c];
}

// ---------------------------------------------------------------------------
extern "C" void kernel_launch(void* const* d_in, const int* in_sizes, int n_in,
                              void* d_out, int out_size, void* d_ws, size_t ws_size,
                              hipStream_t stream) {
    const float* x  = (const float*)d_in[0];
    const float* Wq = (const float*)d_in[1];
    const float* bq = (const float*)d_in[2];
    const float* Wk = (const float*)d_in[3];
    const float* bk = (const float*)d_in[4];
    const float* Wv = (const float*)d_in[5];
    const float* bv = (const float*)d_in[6];
    float* out = (float*)d_out;

    char* ws = (char*)d_ws;
    _Float16* WqH   = (_Float16*)(ws);                    //  0.50 MB
    _Float16* WkH   = (_Float16*)(ws +  524288);          //  0.50 MB
    _Float16* WvH   = (_Float16*)(ws + 1048576);          //  0.50 MB
    _Float16* XavgH = (_Float16*)(ws + 1572864);          //  2 MB (2048x512)
    _Float16* XsumH = (_Float16*)(ws + 3670016);          //  2 MB
    _Float16* Qh    = (_Float16*)(ws + 5767168);          //  2 MB
    _Float16* Kh    = (_Float16*)(ws + 7864320);          //  2 MB
    _Float16* VsumT = (_Float16*)(ws + 9961472);          //  2 MB (8 x 512 x 256)
    float*    scoresF = (float*)(ws + 12058624);          //  2 MB (8 x 256 x 256)
    _Float16* AsoftH  = (_Float16*)(ws + 14155776);       //  1 MB
    float*    outSmall = (float*)(ws + 15204352);         //  4 MB (8 x 256 x 512)
    // total ~18.5 MB of d_ws

    // 0) weights -> f16
    cvt_f16_kernel<<<1024, 256, 0, stream>>>(Wq, WqH, 512 * 512);
    cvt_f16_kernel<<<1024, 256, 0, stream>>>(Wk, WkH, 512 * 512);
    cvt_f16_kernel<<<1024, 256, 0, stream>>>(Wv, WvH, 512 * 512);

    // 1) fused reductions of x (single pass over the 67MB tensor)
    reduce_kernel<<<8 * 512, 256, 0, stream>>>(x, XavgH, XsumH);

    // 2) projections: Q, K (row-major), Vsum (transposed, bias*16)
    {
        dim3 g(128, 32);
        qkv_gemm_kernel<<<g, 32, 0, stream>>>(XavgH, WqH, bq, 1.0f,  Qh,    0);
        qkv_gemm_kernel<<<g, 32, 0, stream>>>(XavgH, WkH, bk, 1.0f,  Kh,    0);
        qkv_gemm_kernel<<<g, 32, 0, stream>>>(XsumH, WvH, bv, 16.0f, VsumT, 1);
    }

    // 3) attention scores QK^T / sqrt(C)
    {
        dim3 g(16, 16, 8);
        scores_kernel<<<g, 32, 0, stream>>>(Qh, Kh, scoresF);
    }

    // 4) softmax (wave per row)
    softmax_kernel<<<256, 256, 0, stream>>>(scoresF, AsoftH);

    // 5) out_small = A @ Vsum
    {
        dim3 g(16, 32, 8);
        av_kernel<<<g, 32, 0, stream>>>(AsoftH, VsumT, outSmall);
    }

    // 6) broadcast to full (B,C,H,W) output (67MB write, streaming)
    expand_kernel<<<65536, 256, 0, stream>>>(outSmall, out);
}
